// CellListComputer_68289980006641
// MI455X (gfx1250) — matrix operand order repaired and checked by reading, not compiled, not run on hardware
//
#include <hip/hip_runtime.h>

typedef __attribute__((ext_vector_type(16))) _Float16 v16h;
typedef __attribute__((ext_vector_type(8)))  float    v8f;

#define GDIM   38
#define TOTALB (GDIM * GDIM * GDIM)   // 54872

// ---------------------------------------------------------------------------
// Zero the bucket-count region (d_out is poisoned 0xAA by the harness).
// ---------------------------------------------------------------------------
__global__ void cl_zero(int* __restrict__ cnt, int n) {
  int i = blockIdx.x * blockDim.x + threadIdx.x;
  if (i < n) cnt[i] = 0;
}

// ---------------------------------------------------------------------------
// Boundary-case table, computed arithmetically. Per axis:
//   class 0 = padded coord 0 (wrapped low), 1 = interior, 2 = padded coord G+1.
// The 27 entries (values 0..17, 5 bits each) are packed into immediates.
// ---------------------------------------------------------------------------
__device__ __forceinline__ int tcase_lookup(int cx, int cy, int cz) {
  int idx = cx * 9 + cy * 3 + cz;
  // entries  0..11: 11,2,0,8,1,0,5,14,0,12,3,0
  const unsigned long long P0 =
      11ull | (2ull << 5) | (8ull << 15) | (1ull << 20) | (5ull << 30) |
      (14ull << 35) | (12ull << 45) | (3ull << 50);
  // entries 12..23: 9,0,0,6,15,0,13,4,0,10,17,0
  const unsigned long long P1 =
      9ull | (6ull << 15) | (15ull << 20) | (13ull << 30) | (4ull << 35) |
      (10ull << 45) | (17ull << 50);
  // entries 24..26: 7,16,0
  const unsigned long long P2 = 7ull | (16ull << 5);
  unsigned long long p = (idx < 12) ? P0 : ((idx < 24) ? P1 : P2);
  int sh = (idx < 12) ? idx : ((idx < 24) ? idx - 12 : idx - 24);
  return (int)((p >> (5 * sh)) & 31ull);
}

// ---------------------------------------------------------------------------
// Fused per-atom pass: frac, flat bucket index, 13 neighbor flat indices +
// translation cases (all in registers, no table gathers), histogram atomics.
// ---------------------------------------------------------------------------
__global__ void cl_main(const float* __restrict__ coords,
                        const float* __restrict__ cell,
                        float* __restrict__ frac_out,
                        int* __restrict__ afi,
                        int* __restrict__ nf,
                        int* __restrict__ nt,
                        int* __restrict__ cnt,
                        int n) {
  const int i = blockIdx.x * blockDim.x + threadIdx.x;
  if (i >= n) return;

  const float cd0 = cell[0], cd1 = cell[4], cd2 = cell[8];  // uniform s_loads

  float x = coords[3 * i + 0];
  float y = coords[3 * i + 1];
  float z = coords[3 * i + 2];

  float fx = x / cd0; fx -= floorf(fx);
  if (fx >= 1.0f) fx -= 1.0f; if (fx < 0.0f) fx += 1.0f;
  float fy = y / cd1; fy -= floorf(fy);
  if (fy >= 1.0f) fy -= 1.0f; if (fy < 0.0f) fy += 1.0f;
  float fz = z / cd2; fz -= floorf(fz);
  if (fz >= 1.0f) fz -= 1.0f; if (fz < 0.0f) fz += 1.0f;

  frac_out[3 * i + 0] = fx;
  frac_out[3 * i + 1] = fy;
  frac_out[3 * i + 2] = fz;

  int ix = (int)floorf(fx * (float)GDIM);
  int iy = (int)floorf(fy * (float)GDIM);
  int iz = (int)floorf(fz * (float)GDIM);
  ix = ix < 0 ? 0 : (ix > GDIM - 1 ? GDIM - 1 : ix);
  iy = iy < 0 ? 0 : (iy > GDIM - 1 ? GDIM - 1 : iy);
  iz = iz < 0 ? 0 : (iz > GDIM - 1 ? GDIM - 1 : iz);

  const int flat = (ix * GDIM + iy) * GDIM + iz;
  afi[i] = flat;
  atomicAdd(&cnt[flat], 1);

  const int DX[13] = {-1, -1, 0, 1, -1, 0, 1, -1, 0, 1, -1, 0, 1};
  const int DY[13] = {0, -1, -1, -1, 1, 1, 1, 0, 0, 0, -1, -1, -1};
  const int DZ[13] = {0, 0, 0, 0, -1, -1, -1, -1, -1, -1, -1, -1, -1};

#pragma unroll
  for (int d = 0; d < 13; ++d) {
    int ax = ix + DX[d], ay = iy + DY[d], az = iz + DZ[d];
    int cx = (ax < 0) ? 0 : ((ax >= GDIM) ? 2 : 1);
    int cy = (ay < 0) ? 0 : ((ay >= GDIM) ? 2 : 1);
    int cz = (az < 0) ? 0 : ((az >= GDIM) ? 2 : 1);
    int wx = ax + ((ax < 0) ? GDIM : 0) - ((ax >= GDIM) ? GDIM : 0);
    int wy = ay + ((ay < 0) ? GDIM : 0) - ((ay >= GDIM) ? GDIM : 0);
    int wz = az + ((az < 0) ? GDIM : 0) - ((az >= GDIM) ? GDIM : 0);
    nf[13 * i + d] = (wx * GDIM + wy) * GDIM + wz;
    nt[13 * i + d] = tcase_lookup(cx, cy, cz);
  }
}

// Uniform (broadcast) lane read: v_readlane_b32, no LDS traffic.
__device__ __forceinline__ float rlane(float v, int l) {
  return __uint_as_float(__builtin_amdgcn_readlane(__float_as_uint(v), l));
}

// ---------------------------------------------------------------------------
// Exclusive prefix sum of `cnt` via WMMA triangular-matrix matmuls.
// One 256-thread block (8 wave32s). Each wave scans a 256-element tile
// (a 16x16 matrix M) with two v_wmma_f32_16x16x32_f16:
//   P = M x U          (U upper-triangular ones -> row-inclusive prefix)
//   S = Lstrict x R + P (R = broadcast rowsums -> cross-row offsets)
// Counts < 2048 are exact in f16; accumulation is f32-exact (< 2^24).
// All OOB handling is clamp-address + select (no divergent branchy loads).
// ---------------------------------------------------------------------------
__global__ void cl_scan(const int* __restrict__ cnt, int* __restrict__ ccnt,
                        int total) {
  __shared__ float wtot[8];
  const int lane = threadIdx.x & 31;
  const int wv = threadIdx.x >> 5;
  float carry = 0.0f;
  const int nchunks = (total + 2047) / 2048;

  // Loop-invariant WMMA operands, built once and kept in VGPRs.
  // B operand: upper-triangular ones U (K<=col), column = lane&15, table K order
  v16h b, a2;
#pragma unroll
  for (int j = 0; j < 16; ++j)
    b[j] = (_Float16)(((j < 8 ? ((lane >> 4) << 3) + j : 16) <= (lane & 15)) ? 1.0f : 0.0f);
  // A2 = strictly-lower triangular ones Lstrict (K < row)
#pragma unroll
  for (int j = 0; j < 8; ++j) {
    int K = (lane < 16) ? j : (8 + j);
    int r = lane & 15;
    a2[j] = (_Float16)((K < r) ? 1.0f : 0.0f);
    a2[j + 8] = (_Float16)0.0f;
  }

  for (int ch = 0; ch < nchunks; ++ch) {
    const int base = ch * 2048 + wv * 256;

    // A operand: 16x16 tile M embedded in the 16x32 f16 A layout (K 16..31 = 0)
    // lane r  (r<16): row r, K 0..7  -> M[r][0..7]  = v[r*16 + 0..7]
    // lane r+16     : row r, K 8..15 -> M[r][8..15] = v[r*16 + 8..15]
    v16h a;
    const int start = base + (lane & 15) * 16 + (lane >> 4) * 8;
#pragma unroll
    for (int j = 0; j < 8; ++j) {
      int g = start + j;
      int gc = (g < total) ? g : (total - 1);  // clamped: unconditional load
      int v = cnt[gc];
      v = (g < total) ? v : 0;
      a[j] = (_Float16)(float)v;
      a[j + 8] = (_Float16)0.0f;
    }

    v8f p = {};
    p = __builtin_amdgcn_wmma_f32_16x16x32_f16(false, a, false, b, (short)0, p,
                                               false, false);

    // Row sums live in column 15: lanes 15 (rows 0..7) and 31 (rows 8..15).
    // Uniform per wave -> v_readlane, not ds_bpermute.
    float rs[16];
#pragma unroll
    for (int d = 0; d < 8; ++d) {
      rs[d] = rlane(p[d], 15);
      rs[d + 8] = rlane(p[d], 31);
    }

    // B2 = rowsums broadcast per column (table K order: K = 8*(lane>=16) + j, j<8).
    v16h b2;
#pragma unroll
    for (int j = 0; j < 16; ++j)
      b2[j] = (j < 8) ? (_Float16)((lane < 16) ? rs[j] : rs[(j + 8) & 15]) : (_Float16)0.0f;

    // S = Lstrict x R + P  (inclusive flat scan of the 256-element tile)
    v8f s = __builtin_amdgcn_wmma_f32_16x16x32_f16(false, a2, false, b2,
                                                   (short)0, p, false, false);

    // Cross-wave combine: tile total = S[15][15] (lane 31, d=7).
    float mytot = rlane(s[7], 31);
    if (lane == 0) wtot[wv] = mytot;
    __syncthreads();

    float prefix = carry;
    float allsum = 0.0f;
#pragma unroll
    for (int w = 0; w < 8; ++w) {
      float t = wtot[w];
      if (w < wv) prefix += t;
      allsum += t;
    }

    // Emit exclusive scan: e[g] = inclusive[g] - v[g] + prefix.
    // D layout: lane holds column c = lane&15, rows r = d + 8*(lane>=16).
#pragma unroll
    for (int d = 0; d < 8; ++d) {
      int r = d + ((lane >> 4) << 3);
      int c = lane & 15;
      int g = base + r * 16 + c;
      int gc = (g < total) ? g : (total - 1);  // clamped: unconditional load
      float val = (float)cnt[gc];
      float excl = s[d] - val + prefix;
      if (g < total) ccnt[g] = (int)(excl + 0.5f);
    }
    carry += allsum;
    __syncthreads();
  }
}

// ---------------------------------------------------------------------------
// Launcher. Output layout (flat, return order):
//   frac (3N f32) | atom_flat (N i32) | neighbor_flat (13N i32) |
//   neighbor_trans (13N i32) | count (54872 i32) | cumcount (54872 i32)
// ---------------------------------------------------------------------------
extern "C" void kernel_launch(void* const* d_in, const int* in_sizes, int n_in,
                              void* d_out, int out_size, void* d_ws,
                              size_t ws_size, hipStream_t stream) {
  const float* coords = (const float*)d_in[0];
  const float* cell = (const float*)d_in[1];
  const int N = in_sizes[0] / 3;

  float* out = (float*)d_out;
  float* frac = out;
  int* afi = (int*)(out + (size_t)3 * N);
  int* nf = (int*)(out + (size_t)4 * N);
  int* nt = (int*)(out + (size_t)17 * N);
  int* cnt = (int*)(out + (size_t)30 * N);
  int* ccnt = cnt + TOTALB;

  cl_zero<<<(TOTALB + 255) / 256, 256, 0, stream>>>(cnt, TOTALB);
  cl_main<<<(N + 255) / 256, 256, 0, stream>>>(coords, cell, frac, afi, nf, nt,
                                               cnt, N);
  cl_scan<<<1, 256, 0, stream>>>(cnt, ccnt, TOTALB);
}
